// Layer_30442728194568
// MI455X (gfx1250) — matrix operand order repaired and checked by reading, not compiled, not run on hardware
//
#include <hip/hip_runtime.h>

typedef __attribute__((ext_vector_type(16))) _Float16 v16h;
typedef __attribute__((ext_vector_type(8)))  _Float16 h8v;
typedef __attribute__((ext_vector_type(2)))  _Float16 h2;
typedef __attribute__((ext_vector_type(8)))  float    v8f;

#define NBLADES 8
#define BATCH   4096
#define FDIM    512
#define KDIM    512
#define NPATH   20

// ---------------------------------------------------------------------------
// Compile-time Cayley table for Cl(3,0), canonical order [1,e1,e2,e3,e12,e13,e23,e123]
// ---------------------------------------------------------------------------
struct Cayley { int ti[64]; int tj[64]; int tk[64]; int tp[64]; float ts[64]; };

constexpr int popc3(int v) { return ((v >> 0) & 1) + ((v >> 1) & 1) + ((v >> 2) & 1); }

constexpr Cayley build_cayley() {
  Cayley c = {};
  int masks[8] = {};
  int idx = 0;
  for (int g = 0; g < 4; ++g)
    for (int m = 0; m < 8; ++m)
      if (popc3(m) == g) masks[idx++] = m;     // sorted by (grade, mask)
  int m2i[8] = {};
  for (int i = 0; i < 8; ++i) m2i[masks[i]] = i;
  int trip[64] = {};
  for (int i = 0; i < 8; ++i) {
    for (int j = 0; j < 8; ++j) {
      const int a = masks[i], b = masks[j];
      int swaps = 0;
      for (int bit = 0; bit < 3; ++bit)
        if ((b >> bit) & 1) swaps += popc3(a >> (bit + 1));
      const int t = i * 8 + j;
      c.ti[t] = i; c.tj[t] = j; c.tk[t] = m2i[a ^ b];
      c.ts[t] = (swaps & 1) ? -1.0f : 1.0f;
      trip[t] = popc3(masks[i]) * 16 + popc3(masks[j]) * 4 + popc3(masks[c.tk[t]]);
    }
  }
  int uniq[64] = {}; int nu = 0;               // sorted unique grade triples (20)
  for (int v = 0; v < 64; ++v) {
    bool present = false;
    for (int t = 0; t < 64; ++t) if (trip[t] == v) present = true;
    if (present) uniq[nu++] = v;
  }
  for (int t = 0; t < 64; ++t)
    for (int p = 0; p < nu; ++p)
      if (uniq[p] == trip[t]) { c.tp[t] = p; break; }
  return c;
}

constexpr Cayley CAY = build_cayley();

// ---------------------------------------------------------------------------
// Fast tanh-approx GELU: hardware v_tanh_f32 if available, else v_exp_f32 path
// ---------------------------------------------------------------------------
__device__ __forceinline__ float gelu_tanh(float v) {
  const float u = 0.7978845608028654f * (v + 0.044715f * v * v * v);
#if __has_builtin(__builtin_amdgcn_tanhf)
  const float th = __builtin_amdgcn_tanhf(u);
#else
  const float e  = __expf(2.0f * u);          // v_exp_f32, branch-free
  const float th = 1.0f - 2.0f / (e + 1.0f);  // graceful at +-inf
#endif
  return 0.5f * v * (1.0f + th);
}

// ---------------------------------------------------------------------------
// Kernel 1: blade-wise GEMM (f16 hi/lo split, 3x v_wmma_f32_16x16x32_f16) + GELU
// Software-pipelined: next tile's global loads land in registers during WMMAs,
// convert+LDS-store after WMMAs into the other LDS buffer; 1 barrier/iter.
// ---------------------------------------------------------------------------
#define BM 128
#define BN 64
#define BK 32
#define LDA 40   // halves; 80B row stride = 20 banks; 16B-aligned K-runs, conflict-free
#define LDB 40

__global__ __launch_bounds__(256) void gemm_gelu_kernel(
    const float* __restrict__ X,    // [8, 4096, 512]
    const float* __restrict__ Wt,   // [4, 512, 512]
    const float* __restrict__ bias, // [512]
    float* __restrict__ Y)          // [8, 4096, 512]  (= d_out, y buffer)
{
  __shared__ __align__(16) _Float16 AhS[2][BM * LDA];
  __shared__ __align__(16) _Float16 AlS[2][BM * LDA];
  __shared__ __align__(16) _Float16 BhS[2][BN * LDB];
  __shared__ __align__(16) _Float16 BlS[2][BN * LDB];

  const int blade = blockIdx.z;
  const int gsel  = (blade == 0) ? 0 : (blade <= 3 ? 1 : (blade <= 6 ? 2 : 3));
  const int n0 = blockIdx.x * BN;
  const int m0 = blockIdx.y * BM;
  const float* Xb = X  + (size_t)blade * BATCH * KDIM;
  const float* Wg = Wt + (size_t)gsel  * KDIM * FDIM;

  const int t    = threadIdx.x;
  const int wid  = t >> 5;
  const int lane = t & 31;
  const int wm = (wid & 3) * 32;   // wave tile row within block tile
  const int wn = (wid >> 2) * 32;  // wave tile col within block tile
  const int lh = lane & 15;
  const bool hiLane = lane >= 16;

  // staging registers for the in-flight tile (A: 8 float2 pairs, B: 8 floats)
  float a0[8], a1[8], bst[8];

  auto loadTile = [&](int k0) {
    #pragma unroll
    for (int i = 0; i < 8; ++i) {                 // A: 2048 pairs / 256 thr
      const int e2 = i * 256 + t;
      const int m  = e2 >> 4;
      const int kp = (e2 & 15) * 2;
      const float2 v = *(const float2*)&Xb[(size_t)(m0 + m) * KDIM + (k0 + kp)];
      a0[i] = v.x; a1[i] = v.y;
    }
    #pragma unroll
    for (int i = 0; i < 8; ++i) {                 // B: 2048 elems / 256 thr
      const int e = i * 256 + t;
      const int kk = e >> 6, n = e & 63;
      bst[i] = Wg[(size_t)(k0 + kk) * FDIM + (n0 + n)];
    }
  };

  auto storeTile = [&](int buf) {
    #pragma unroll
    for (int i = 0; i < 8; ++i) {
      const int e2 = i * 256 + t;
      const int m  = e2 >> 4;
      const int kp = (e2 & 15) * 2;
      const _Float16 h0 = (_Float16)a0[i], h1 = (_Float16)a1[i];
      h2 hh; hh.x = h0; hh.y = h1;
      h2 ll; ll.x = (_Float16)(a0[i] - (float)h0); ll.y = (_Float16)(a1[i] - (float)h1);
      *(h2*)&AhS[buf][m * LDA + kp] = hh;
      *(h2*)&AlS[buf][m * LDA + kp] = ll;
    }
    #pragma unroll
    for (int i = 0; i < 8; ++i) {
      const int e = i * 256 + t;
      const int kk = e >> 6, n = e & 63;
      const _Float16 h = (_Float16)bst[i];
      BhS[buf][n * LDB + kk] = h;
      BlS[buf][n * LDB + kk] = (_Float16)(bst[i] - (float)h);
    }
  };

  const v8f vzero = {0.f, 0.f, 0.f, 0.f, 0.f, 0.f, 0.f, 0.f};
  v8f acc[2][2];
  acc[0][0] = vzero; acc[0][1] = vzero; acc[1][0] = vzero; acc[1][1] = vzero;

  // prologue: stage tile 0
  loadTile(0);
  storeTile(0);
  __syncthreads();

  const int NIT = KDIM / BK;   // 16
  for (int it = 0; it < NIT; ++it) {
    const int cur = it & 1;

    // gather WMMA fragments (ISA 7.12.2 16-bit layouts); each = 2x ds_load_b128
    // A lane<16: halves 0-7 -> K0..7, 8-15 -> K16..23 ; lane>=16: +8
    // B lane<16: halves 0-7 -> K0..7, 8-15 -> K16..23 ; lane>=16: +8
    v16h afh[2], afl[2], bfh[2], bfl[2];
    const int aoff = hiLane ? 8 : 0;
    const int boff = hiLane ? 8 : 0;
    #pragma unroll
    for (int mi = 0; mi < 2; ++mi) {
      const int mbase = (wm + mi * 16 + lh) * LDA;
      const h8v ah0 = *(const h8v*)&AhS[cur][mbase + aoff];
      const h8v ah1 = *(const h8v*)&AhS[cur][mbase + 16 + aoff];
      const h8v al0 = *(const h8v*)&AlS[cur][mbase + aoff];
      const h8v al1 = *(const h8v*)&AlS[cur][mbase + 16 + aoff];
      afh[mi] = __builtin_shufflevector(ah0, ah1, 0,1,2,3,4,5,6,7,8,9,10,11,12,13,14,15);
      afl[mi] = __builtin_shufflevector(al0, al1, 0,1,2,3,4,5,6,7,8,9,10,11,12,13,14,15);
    }
    #pragma unroll
    for (int ni = 0; ni < 2; ++ni) {
      const int nbase = (wn + ni * 16 + lh) * LDB;
      const h8v bh0 = *(const h8v*)&BhS[cur][nbase + boff];
      const h8v bh1 = *(const h8v*)&BhS[cur][nbase + boff + 16];
      const h8v bl0 = *(const h8v*)&BlS[cur][nbase + boff];
      const h8v bl1 = *(const h8v*)&BlS[cur][nbase + boff + 16];
      bfh[ni] = __builtin_shufflevector(bh0, bh1, 0,1,2,3,4,5,6,7,8,9,10,11,12,13,14,15);
      bfl[ni] = __builtin_shufflevector(bl0, bl1, 0,1,2,3,4,5,6,7,8,9,10,11,12,13,14,15);
    }

    // issue next tile's global loads now; data arrives while WMMAs run
    const bool has_next = (it + 1) < NIT;
    if (has_next) loadTile((it + 1) * BK);

    // D = Ah*Bh + Ah*Bl + Al*Bh + C   (independent accumulators back-to-back)
    #pragma unroll
    for (int mi = 0; mi < 2; ++mi)
      #pragma unroll
      for (int ni = 0; ni < 2; ++ni)
        acc[mi][ni] = __builtin_amdgcn_wmma_f32_16x16x32_f16(
            false, afh[mi], false, bfh[ni], (short)0, acc[mi][ni], false, false);
    #pragma unroll
    for (int mi = 0; mi < 2; ++mi)
      #pragma unroll
      for (int ni = 0; ni < 2; ++ni)
        acc[mi][ni] = __builtin_amdgcn_wmma_f32_16x16x32_f16(
            false, afh[mi], false, bfl[ni], (short)0, acc[mi][ni], false, false);
    #pragma unroll
    for (int mi = 0; mi < 2; ++mi)
      #pragma unroll
      for (int ni = 0; ni < 2; ++ni)
        acc[mi][ni] = __builtin_amdgcn_wmma_f32_16x16x32_f16(
            false, afl[mi], false, bfh[ni], (short)0, acc[mi][ni], false, false);

    // convert + store next tile into the other buffer, then sync
    if (has_next) storeTile(cur ^ 1);
    __syncthreads();
  }

  // epilogue: bias + GELU, store y
  #pragma unroll
  for (int mi = 0; mi < 2; ++mi) {
    #pragma unroll
    for (int ni = 0; ni < 2; ++ni) {
      const int ncol = n0 + wn + ni * 16 + lh;
      const float bv = bias[ncol];
      #pragma unroll
      for (int r = 0; r < 8; ++r) {
        const int mrow = m0 + wm + mi * 16 + r + (hiLane ? 8 : 0);
        Y[((size_t)blade * BATCH + mrow) * FDIM + ncol] = gelu_tanh(acc[mi][ni][r] + bv);
      }
    }
  }
}

// ---------------------------------------------------------------------------
// Kernel 2: weighted geometric product + grade-wise RMS norm (in-place on Y)
// One block per sample b; each thread owns 2 consecutive features.
// ---------------------------------------------------------------------------
__global__ __launch_bounds__(256) void gp_norm_kernel(
    const float* __restrict__ X,    // [8, 4096, 512]
    const float* __restrict__ gpw,  // [512, 20]
    float* __restrict__ Y)          // in: y (gelu output); out: result
{
  const int b    = blockIdx.x;
  const int t    = threadIdx.x;
  const int lane = t & 31;
  const int wid  = t >> 5;
  const int f0   = t * 2;
  __shared__ float red[4][8];

  float xv[8][2], yv[8][2];
  #pragma unroll
  for (int i = 0; i < 8; ++i) {
    const size_t off = ((size_t)i * BATCH + b) * FDIM + f0;
    const float2 vx = *(const float2*)&X[off];
    const float2 vy = *(const float2*)&Y[off];   // all reads precede barrier/writes
    xv[i][0] = vx.x; xv[i][1] = vx.y;
    yv[i][0] = vy.x; yv[i][1] = vy.y;
  }
  float wbuf[40];                                 // gpw rows f0, f0+1 (contiguous)
  #pragma unroll
  for (int q = 0; q < 10; ++q) {
    const float4 v = ((const float4*)&gpw[(size_t)f0 * NPATH])[q];
    wbuf[4*q] = v.x; wbuf[4*q+1] = v.y; wbuf[4*q+2] = v.z; wbuf[4*q+3] = v.w;
  }

  float outv[8][2];
  float ssq[4] = {0.f, 0.f, 0.f, 0.f};
  #pragma unroll
  for (int sub = 0; sub < 2; ++sub) {
    const float* w = &wbuf[sub * NPATH];
    float a[8] = {0.f, 0.f, 0.f, 0.f, 0.f, 0.f, 0.f, 0.f};
    #pragma unroll
    for (int tt = 0; tt < 64; ++tt) {             // constexpr indices -> folded FMAs
      a[CAY.tk[tt]] += CAY.ts[tt] * w[CAY.tp[tt]] * xv[CAY.ti[tt]][sub] * yv[CAY.tj[tt]][sub];
    }
    ssq[0] += a[0]*a[0];
    ssq[1] += a[1]*a[1] + a[2]*a[2] + a[3]*a[3];
    ssq[2] += a[4]*a[4] + a[5]*a[5] + a[6]*a[6];
    ssq[3] += a[7]*a[7];
    #pragma unroll
    for (int k = 0; k < 8; ++k) outv[k][sub] = a[k];
  }

  // wave32 reduction, then cross-wave via LDS
  #pragma unroll
  for (int g = 0; g < 4; ++g) {
    float v = ssq[g];
    #pragma unroll
    for (int off = 16; off >= 1; off >>= 1) v += __shfl_xor(v, off, 32);
    if (lane == 0) red[g][wid] = v;
  }
  __syncthreads();

  float inv[4];
  #pragma unroll
  for (int g = 0; g < 4; ++g) {
    float s = 0.f;
    #pragma unroll
    for (int wv = 0; wv < 8; ++wv) s += red[g][wv];
    const float cnt = (g == 0 || g == 3) ? (float)FDIM : 3.0f * (float)FDIM;
    inv[g] = rsqrtf(s / cnt + 1e-6f);
  }

  const int kg[8] = {0, 1, 1, 1, 2, 2, 2, 3};
  #pragma unroll
  for (int k = 0; k < 8; ++k) {
    float2 o;
    o.x = outv[k][0] * inv[kg[k]];
    o.y = outv[k][1] * inv[kg[k]];
    *(float2*)&Y[((size_t)k * BATCH + b) * FDIM + f0] = o;
  }
}

// ---------------------------------------------------------------------------
extern "C" void kernel_launch(void* const* d_in, const int* in_sizes, int n_in,
                              void* d_out, int out_size, void* d_ws, size_t ws_size,
                              hipStream_t stream) {
  (void)in_sizes; (void)n_in; (void)d_ws; (void)ws_size; (void)out_size;
  const float* x   = (const float*)d_in[0];  // [8,4096,512]
  const float* lw  = (const float*)d_in[1];  // [4,512,512]
  const float* lb  = (const float*)d_in[2];  // [512]
  const float* gpw = (const float*)d_in[3];  // [512,20]
  float* out = (float*)d_out;                // [8,4096,512]

  dim3 g1(FDIM / BN, BATCH / BM, NBLADES);   // (8, 32, 8)
  gemm_gelu_kernel<<<g1, 256, 0, stream>>>(x, lw, lb, out);
  gp_norm_kernel<<<BATCH, 256, 0, stream>>>(x, gpw, out);
}
